// GNAE_ENC_78580721647689
// MI455X (gfx1250) — matrix operand order repaired and checked by reading, not compiled, not run on hardware
//
#include <hip/hip_runtime.h>
#include <hip/hip_bf16.h>

typedef __attribute__((ext_vector_type(16))) __bf16        v16bf;
typedef __attribute__((ext_vector_type(8)))  float         v8f;
typedef __attribute__((ext_vector_type(8)))  unsigned int  v8u;
typedef __attribute__((ext_vector_type(4)))  unsigned int  u32x4;
typedef __attribute__((ext_vector_type(8)))  int           i32x8;
typedef __attribute__((ext_vector_type(4)))  int           i32x4;

#define NNODES 10000
#define NEDGES 160000
#define FIN    512
#define FH     512
#define FOUT   256
#define KK     512              // inner (reduction) dim: 512 for every layer
#define MTILES (NNODES / 16)    // 625
#define BPAD   8                // bf16 elems of LDS row padding (16B) -> bank spread
#define BSTRIDE (KK + BPAD)

// ---------------- utility kernels ----------------

__global__ void zero_f32(float* __restrict__ p, long long n) {
  long long i = (long long)blockIdx.x * blockDim.x + threadIdx.x;
  if (i < n) p[i] = 0.0f;
}

__global__ void count_deg(const int* __restrict__ col, float* __restrict__ deg, int E) {
  int i = blockIdx.x * blockDim.x + threadIdx.x;
  if (i < E) atomicAdd(&deg[col[i]], 1.0f);
}

// deg counted over edges only; +1.0 folds in the self-loop.
__global__ void make_dinv(const float* __restrict__ deg, float* __restrict__ dinv, int n) {
  int i = blockIdx.x * blockDim.x + threadIdx.x;
  if (i < n) dinv[i] = rsqrtf(deg[i] + 1.0f);
}

__global__ void f32_to_bf16(const float* __restrict__ src, __bf16* __restrict__ dst, long long n) {
  long long i = (long long)blockIdx.x * blockDim.x + threadIdx.x;
  if (i < n) dst[i] = (__bf16)src[i];
}

// Wt[n*K + k] = W[k*Nc + n]  (transpose to column-major so B fragments load contiguously)
__global__ void transpose_to_bf16(const float* __restrict__ W, __bf16* __restrict__ Wt,
                                  int K, int Nc) {
  int i = blockIdx.x * blockDim.x + threadIdx.x;
  if (i < K * Nc) {
    int k = i / Nc, n = i % Nc;
    Wt[(size_t)n * K + k] = (__bf16)W[i];
  }
}

// ---------------- WMMA bf16 GEMM: C[M,Nc] = A[M,512] * Bt[Nc,512]^T ----------------
// Block = 8 waves. Block tile = 128 rows x 64 cols.
// B panel (64 cols x 512 K, bf16) staged in LDS once per block via the Tensor Data
// Mover (TDM) with D# row padding; each wave computes a 16x64 strip (4 accumulators),
// reusing each A fragment across 4 v_wmma_f32_16x16x32_bf16.
__global__ void __launch_bounds__(256)
gemm_bf16_wmma(const __bf16* __restrict__ A,   // [M,512] row-major
               const __bf16* __restrict__ Bt,  // [Nc,512] row-major
               float* __restrict__ C,          // [M,Nc] row-major
               int M, int Nc) {
  __shared__ __bf16 Bs[64 * BSTRIDE];          // 65 KB padded panel

  const int lane = threadIdx.x & 31;
  const int wave = threadIdx.x >> 5;
  const int l15  = lane & 15;
  const int kg   = (lane >> 4) & 1;            // lane group: 0 = lanes 0-15, 1 = 16-31
  const int ncol0 = blockIdx.x * 64;           // first column of this block's panel

  // ---- stage B panel: Bt rows [ncol0, ncol0+63], full K, into padded LDS ----
#if __has_builtin(__builtin_amdgcn_tensor_load_to_lds)
  if (wave == 0) {
    unsigned lds_base = (unsigned)(unsigned long long)(const void*)&Bs[0];
    unsigned long long ga =
        (unsigned long long)(const void*)(Bt + (size_t)ncol0 * KK);
    u32x4 g0;
    g0[0] = 1u;                                   // count=1 (valid descriptor)
    g0[1] = lds_base;                             // lds_addr
    g0[2] = (unsigned)(ga & 0xffffffffu);         // global_addr[31:0]
    g0[3] = (unsigned)(ga >> 32) | (2u << 30);    // global_addr[56:32] | type=2
    i32x8 g1;
    // data_size=1 (2B) | pad_enable | pad_interval=7 (256 dw = one 512-elem row)
    // | pad_amount=3 (4 dwords = 16B = BPAD elems)
    g1[0] = (1 << 16) | (1 << 20) | (7 << 22) | (3 << 25);
    g1[1] = (KK & 0xffff) << 16;                  // tensor_dim0[15:0]
    g1[2] = (KK >> 16) | (64 << 16);              // tensor_dim0[31:16] | tensor_dim1
    g1[3] = (KK & 0xffff) << 16;                  // tile_dim0 = KK
    g1[4] = 64;                                   // tile_dim1 = 64, tile_dim2 = 0
    g1[5] = KK;                                   // tensor_dim0_stride[31:0]
    g1[6] = 0;
    g1[7] = 0;
    i32x4 gz4 = {0, 0, 0, 0};                     // 2D tensor: groups 2/3 disabled
    i32x8 gz8 = {0, 0, 0, 0, 0, 0, 0, 0};
    __builtin_amdgcn_tensor_load_to_lds(g0, g1, gz4, gz4, gz8, 0);
    __builtin_amdgcn_s_wait_tensorcnt(0);
  }
  __syncthreads();
#else
  // cooperative fallback: 16B per thread per iteration into the padded layout
  for (int idx = threadIdx.x; idx < 64 * (KK / 8); idx += blockDim.x) {
    int r  = idx / (KK / 8);
    int kk = (idx % (KK / 8)) * 8;
    *(uint4*)&Bs[r * BSTRIDE + kk] =
        *(const uint4*)&Bt[(size_t)(ncol0 + r) * KK + kk];
  }
  __syncthreads();
#endif

  // ---- compute: one 16x64 strip per wave ----
  const int mt = blockIdx.y * 8 + wave;
  if (mt < (M >> 4)) {                           // wave-uniform: EXEC all-1s for WMMA
    const __bf16* Arow = A + (size_t)(mt * 16 + l15) * KK;

    v8f c0 = {}, c1 = {}, c2 = {}, c3 = {};
    for (int k0 = 0; k0 < KK; k0 += 32) {
      // A 16x32 fragment: lanes 0-15 hold K {0..15} of row l15,
      // lanes 16-31 hold K {16..31}.  Dwords [8kg..8kg+7].
      const unsigned int* A32 = (const unsigned int*)(Arow + k0);
      v8u au;
#pragma unroll
      for (int j = 0; j < 4; ++j) {
        au[j]     = A32[j + 8 * kg];
        au[j + 4] = A32[j + 4 + 8 * kg];
      }
      v16bf a = __builtin_bit_cast(v16bf, au);
      if (k0 + 32 < KK) __builtin_prefetch(A32 + 16, 0, 0);  // global_prefetch_b8

      // 4 B fragments from the padded LDS panel; rows are 16B-aligned (1040B stride)
#pragma unroll
      for (int j = 0; j < 4; ++j) {
        const __bf16* bp = &Bs[(j * 16 + l15) * BSTRIDE + k0 + (kg << 4)];
        union { uint4 q[2]; v16bf v; } ub;
        ub.q[0] = *(const uint4*)bp;
        ub.q[1] = *(const uint4*)(bp + 8);
        v8f& acc = (j == 0) ? c0 : (j == 1) ? c1 : (j == 2) ? c2 : c3;
        acc = __builtin_amdgcn_wmma_f32_16x16x32_bf16(
            false, a, false, ub.v, (short)0, acc, false, false);
      }
    }

    // D layout: VGPR v -> row (v + 8*kg), col = l15 (+ 16 per n-subtile)
    float* Cb = C + (size_t)(mt * 16 + kg * 8) * Nc + ncol0 + l15;
#pragma unroll
    for (int v = 0; v < 8; ++v) {
      Cb[(size_t)v * Nc +  0] = c0[v];
      Cb[(size_t)v * Nc + 16] = c1[v];
      Cb[(size_t)v * Nc + 32] = c2[v];
      Cb[(size_t)v * Nc + 48] = c3[v];
    }
  }
}

// ---------------- message passing ----------------

// one block per edge; blockDim = F/4; atomic scatter-add of xw[row]*norm into out[col]
__global__ void scatter_edges(const float* __restrict__ XW, const int* __restrict__ row,
                              const int* __restrict__ col, const float* __restrict__ dinv,
                              float* __restrict__ out, int F) {
  int e = blockIdx.x;
  int r = row[e], c = col[e];
  float nrm = dinv[r] * dinv[c];
  int f = threadIdx.x * 4;
  const float4 v = *(const float4*)(XW + (size_t)r * F + f);
  float* o = out + (size_t)c * F + f;
  atomicAdd(o + 0, v.x * nrm);
  atomicAdd(o + 1, v.y * nrm);
  atomicAdd(o + 2, v.z * nrm);
  atomicAdd(o + 3, v.w * nrm);
}

// fused self-loop contribution + bias + optional ReLU (runs after scatter, same stream)
__global__ void post_bias(float* __restrict__ H, const float* __restrict__ XW,
                          const float* __restrict__ dinv, const float* __restrict__ b,
                          int F, int relu, long long n) {
  long long i = (long long)blockIdx.x * blockDim.x + threadIdx.x;
  if (i >= n) return;
  int node = (int)(i / F);
  int f    = (int)(i % F);
  float dv = dinv[node];
  float v = H[i] + XW[i] * dv * dv + b[f];
  if (relu) v = fmaxf(v, 0.0f);
  H[i] = v;
}

// row-wise: H = H * (1.5 / max(||H||2, eps)) [+ res]
__global__ void l2norm_rows(float* __restrict__ H, const float* __restrict__ res, int F) {
  int node = blockIdx.x;
  float* rowp = H + (size_t)node * F;
  float ss = 0.0f;
  for (int f = threadIdx.x; f < F; f += blockDim.x) {
    float v = rowp[f];
    ss += v * v;
  }
  __shared__ float red[256];
  red[threadIdx.x] = ss;
  __syncthreads();
  for (int s = blockDim.x >> 1; s > 0; s >>= 1) {
    if (threadIdx.x < (unsigned)s) red[threadIdx.x] += red[threadIdx.x + s];
    __syncthreads();
  }
  float scale = 1.5f / fmaxf(sqrtf(red[0]), 1e-12f);
  for (int f = threadIdx.x; f < F; f += blockDim.x) {
    float v = rowp[f] * scale;
    if (res) v += res[(size_t)node * F + f];
    rowp[f] = v;
  }
}

// ---------------- launch ----------------

extern "C" void kernel_launch(void* const* d_in, const int* in_sizes, int n_in,
                              void* d_out, int out_size, void* d_ws, size_t ws_size,
                              hipStream_t stream) {
  (void)in_sizes; (void)n_in; (void)out_size; (void)ws_size;

  const float* x  = (const float*)d_in[0];
  const int*   ei = (const int*)d_in[1];     // edge_index flat (2,E)
  const int* row  = ei;
  const int* col  = ei + NEDGES;
  const float* W0 = (const float*)d_in[2];  const float* b0 = (const float*)d_in[3];
  const float* W1 = (const float*)d_in[4];  const float* b1 = (const float*)d_in[5];
  const float* W2 = (const float*)d_in[6];  const float* b2 = (const float*)d_in[7];
  const float* Wx = (const float*)d_in[8];  const float* bx = (const float*)d_in[9];
  float* out = (float*)d_out;

  char* wsp = (char*)d_ws;
  auto alloc = [&](size_t bytes) -> char* {
    char* p = wsp;
    wsp += (bytes + 255) & ~(size_t)255;
    return p;
  };
  float*  deg  = (float*)alloc((size_t)NNODES * 4);
  float*  dinv = (float*)alloc((size_t)NNODES * 4);
  __bf16* Xb   = (__bf16*)alloc((size_t)NNODES * FH * 2);
  __bf16* Wt   = (__bf16*)alloc((size_t)FH * FH * 2);
  float*  XW   = (float*)alloc((size_t)NNODES * FH * 4);
  float*  Ha   = (float*)alloc((size_t)NNODES * FH * 4);
  float*  Hb   = (float*)alloc((size_t)NNODES * FH * 4);

  // degree -> dinv
  zero_f32<<<(NNODES + 255) / 256, 256, 0, stream>>>(deg, NNODES);
  count_deg<<<(NEDGES + 255) / 256, 256, 0, stream>>>(col, deg, NEDGES);
  make_dinv<<<(NNODES + 255) / 256, 256, 0, stream>>>(deg, dinv, NNODES);

  auto conv = [&](const float* in_act, const float* W, const float* b,
                  float* dstH, int Fin, int Fo, int relu) {
    long long nin = (long long)NNODES * Fin;
    f32_to_bf16<<<(int)((nin + 255) / 256), 256, 0, stream>>>(in_act, Xb, nin);
    transpose_to_bf16<<<(Fin * Fo + 255) / 256, 256, 0, stream>>>(W, Wt, Fin, Fo);

    dim3 ggrid(Fo / 64, (MTILES + 7) / 8);
    gemm_bf16_wmma<<<ggrid, 256, 0, stream>>>(Xb, Wt, XW, NNODES, Fo);

    long long nout = (long long)NNODES * Fo;
    zero_f32<<<(int)((nout + 255) / 256), 256, 0, stream>>>(dstH, nout);
    scatter_edges<<<NEDGES, Fo / 4, 0, stream>>>(XW, row, col, dinv, dstH, Fo);
    post_bias<<<(int)((nout + 255) / 256), 256, 0, stream>>>(dstH, XW, dinv, b, Fo, relu, nout);
  };

  // layer 0: h0 = relu(conv0(x));  res = Ha
  conv(x, W0, b0, Ha, FIN, FH, 1);
  // layer 1: h1 = l2norm(conv1(h0)) * 1.5 + res
  conv(Ha, W1, b1, Hb, FH, FH, 0);
  l2norm_rows<<<NNODES, 256, 0, stream>>>(Hb, Ha, FH);
  // layer 2: h2 = l2norm(conv2(h1)) * 1.5     (Ha free now)
  conv(Hb, W2, b2, Ha, FH, FH, 0);
  l2norm_rows<<<NNODES, 256, 0, stream>>>(Ha, nullptr, FH);
  // layer 3: out = conv3(h2)
  conv(Ha, Wx, bx, out, FH, FOUT, 0);
}
